// RouteNet_25331717112312
// MI455X (gfx1250) — matrix operand order repaired and checked by reading, not compiled, not run on hardware
//
#include <hip/hip_runtime.h>

typedef _Float16 h16;
typedef _Float16 v16h __attribute__((ext_vector_type(16)));
typedef float    v8f  __attribute__((ext_vector_type(8)));
typedef unsigned int u32x4 __attribute__((ext_vector_type(4)));
typedef int i32x4 __attribute__((ext_vector_type(4)));
typedef int i32x8 __attribute__((ext_vector_type(8)));

#define NB    8192      // batch
#define NIN   3072
#define DD    32
#define NOUT  1000
#define NTOUT 63        // ceil(1000/16)
#define NOUTP (NTOUT*16)
#define BN_EPS 1e-5f

#if __has_builtin(__builtin_amdgcn_tensor_load_to_lds)
#define HAVE_TDM 1
#else
#define HAVE_TDM 0
#endif

static __device__ __forceinline__ v8f vzero8() {
  v8f c;
#pragma unroll
  for (int i = 0; i < 8; i++) c[i] = 0.0f;
  return c;
}

// load a 16-half fragment as two contiguous 16B chunks
static __device__ __forceinline__ v16h load_frag(const h16* p0, const h16* p1) {
  v16h r;
  *reinterpret_cast<int4*>(&r)       = *reinterpret_cast<const int4*>(p0);
  *(reinterpret_cast<int4*>(&r) + 1) = *reinterpret_cast<const int4*>(p1);
  return r;
}

static __device__ __forceinline__ v8f wmma16(v16h a, v16h b, v8f c) {
  return __builtin_amdgcn_wmma_f32_16x16x32_f16(false, a, false, b, (short)0, c, false, false);
}

#if HAVE_TDM
// Tensor Data Mover: 2-D tile load Global->LDS.
// Descriptor per CDNA5 ISA (D# groups 0..3).
// data_size_log: 0=1B 1=2B 2=4B 3=8B; dims/strides in data_size units.
static __device__ __forceinline__ void tdm_load_2d(unsigned lds_addr, const void* gaddr,
                                                   unsigned data_size_log,
                                                   unsigned tile_d0, unsigned tile_d1,
                                                   unsigned long long stride0) {
  unsigned long long ga = (unsigned long long)gaddr;
  u32x4 g0;
  g0[0] = 1u;                                             // count=1, user descriptor
  g0[1] = lds_addr;                                       // LDS byte address
  g0[2] = (unsigned)(ga & 0xFFFFFFFFu);                   // global_addr[31:0]
  g0[3] = (unsigned)((ga >> 32) & 0x01FFFFFFu) | 0x80000000u;  // global_addr[56:32] | type=2
  unsigned long long s1 = stride0 * (unsigned long long)tile_d1;
  i32x8 g1;
  g1[0] = (int)(data_size_log << 16);                     // workgroup_mask=0, data_size
  g1[1] = (int)((tile_d0 & 0xFFFFu) << 16);               // tensor_dim0[15:0] (== tile_d0)
  g1[2] = (int)(((tile_d0 >> 16) & 0xFFFFu) | ((tile_d1 & 0xFFFFu) << 16)); // tdim0 hi | tdim1 lo
  g1[3] = (int)(((tile_d1 >> 16) & 0xFFFFu) | (tile_d0 << 16));             // tdim1 hi | tile_dim0
  g1[4] = (int)(tile_d1 & 0xFFFFu);                       // tile_dim1 | tile_dim2=0
  g1[5] = (int)(stride0 & 0xFFFFFFFFull);                 // tensor_dim0_stride[31:0]
  g1[6] = (int)(((stride0 >> 32) & 0xFFFFull) | ((s1 & 0xFFFFull) << 16));
  g1[7] = (int)((s1 >> 16) & 0xFFFFFFFFull);              // tensor_dim1_stride[47:16]
  i32x4 gz;
  gz[0] = 0; gz[1] = 0; gz[2] = 0; gz[3] = 0;
#if defined(__clang_major__) && __clang_major__ >= 23
  i32x8 gz8;
#pragma unroll
  for (int i = 0; i < 8; i++) gz8[i] = 0;
  __builtin_amdgcn_tensor_load_to_lds(g0, g1, gz, gz, gz8, 0);
#else
  __builtin_amdgcn_tensor_load_to_lds(g0, g1, gz, gz, 0);
#endif
}
#endif

// ---------------- init / BN stats over input ----------------

__global__ void k_zero_tgate(float* tg) {
  int i = blockIdx.x * 256 + threadIdx.x;
  if (i < NB) tg[i] = 0.0f;
}

// 64 blocks x 128 rows; per-column partial sum/sumsq
__global__ void k_colstats(const float* __restrict__ x, float* psum, float* psumsq) {
  int blk = blockIdx.x;
  int r0  = blk * 128;
  for (int c = threadIdx.x; c < NIN; c += 256) {
    float s = 0.f, s2 = 0.f;
    const float* p = x + (size_t)r0 * NIN + c;
    for (int r = 0; r < 128; r++) {
      __builtin_prefetch(p + (size_t)(r + 8) * NIN, 0, 3);
      float v = p[(size_t)r * NIN];
      s += v; s2 += v * v;
    }
    psum[blk * NIN + c] = s;
    psumsq[blk * NIN + c] = s2;
  }
}

__global__ void k_colfinal(const float* psum, const float* psumsq, float* mean, float* rstd) {
  int c = blockIdx.x * 256 + threadIdx.x;
  if (c >= NIN) return;
  float s = 0.f, s2 = 0.f;
  for (int b = 0; b < 64; b++) { s += psum[b * NIN + c]; s2 += psumsq[b * NIN + c]; }
  float m = s / (float)NB;
  float v = s2 / (float)NB - m * m;
  mean[c] = m;
  rstd[c] = rsqrtf(v + BN_EPS);
}

// ---------------- fold BN into W_in, swizzle weights to B-fragment layout ----------------
// B-frag layout (16-bit, 32xK x 16xN tile): lane = (n&15) + ((k>>3)&1 ? 16 : 0), elem = (k&7) + (k>=16 ? 8 : 0),
// stored as frag_base + lane*16 + elem  (each lane = 16 contiguous halves = 32B)

__global__ void k_fold_win(const float* __restrict__ Win, const float* gamma,
                           const float* mean, const float* rstd, h16* WbIn) {
  int idx = blockIdx.x * 256 + threadIdx.x;     // 3072*64
  if (idx >= NIN * 64) return;
  int j = idx >> 6, d = idx & 63;
  float w = rstd[j] * gamma[j] * Win[(d >> 5) * NIN * DD + j * DD + (d & 31)];
  int kk = j >> 5, kr = j & 31, nt = d >> 4;
  int lane = (d & 15) + (((kr >> 3) & 1) ? 16 : 0);
  int el   = (kr & 7) + ((kr >= 16) ? 8 : 0);
  WbIn[((kk * 4 + nt) * 32 + lane) * 16 + el] = (h16)w;
}

__global__ void k_fold_bias(const float* __restrict__ Win, const float* b_in,
                            const float* gamma, const float* beta,
                            const float* mean, const float* rstd, float* biasIn) {
  int d = threadIdx.x;
  if (d >= 64) return;
  int k = d >> 5, dd = d & 31;
  float acc = b_in[d];
  for (int j = 0; j < NIN; j++)
    acc += (beta[j] - mean[j] * rstd[j] * gamma[j]) * Win[k * NIN * DD + j * DD + dd];
  biasIn[d] = acc;
}

__global__ void k_swz_wd(const float* __restrict__ Wd, h16* WdF) {
  int idx = blockIdx.x * 256 + threadIdx.x;     // 54*1024
  if (idx >= 54 * 1024) return;
  int e = idx >> 10, r = (idx >> 5) & 31, n = idx & 31;
  float v = Wd[idx];                            // [e][k=r][n]
  int nt = n >> 4;
  int lane = (n & 15) + (((r >> 3) & 1) ? 16 : 0);
  int el   = (r & 7) + ((r >= 16) ? 8 : 0);
  WdF[e * 1024 + nt * 512 + lane * 16 + el] = (h16)v;
}

__global__ void k_swz_wout(const float* __restrict__ Wout, h16* WoutF) {
  int idx = blockIdx.x * 256 + threadIdx.x;     // 2*32*1008
  if (idx >= 2 * DD * NOUTP) return;
  int kb = idx / (DD * NOUTP);
  int rem = idx % (DD * NOUTP);
  int r = rem / NOUTP, n = rem % NOUTP;
  float v = (n < NOUT) ? Wout[kb * DD * NOUT + r * NOUT + n] : 0.0f;
  int nt = n >> 4;
  int lane = (n & 15) + (((r >> 3) & 1) ? 16 : 0);
  int el   = (r & 7) + ((r >= 16) ? 8 : 0);
  WoutF[kb * (NTOUT * 512) + nt * 512 + lane * 16 + el] = (h16)v;
}

// ---------------- big input GEMM: actsIn = relu(x @ W' + bias') ----------------
// block = 128 threads (4 waves), block tile = 64 rows x 64 cols, K loop 96 x 32.
// TDM double-buffers the A tile (64x32 f32, 2-D) and B chunk (4KB) into LDS.

#define A_BYTES 8192
#define B_OFF   16384
#define B_BYTES 4096

__global__ void k_ingemm(const float* __restrict__ x, const h16* __restrict__ WbIn,
                         const float* __restrict__ biasIn, float* __restrict__ actsIn) {
  int tid = threadIdx.x;
  int lane = tid & 31, wave = tid >> 5;
  int rows = (blockIdx.x * 4 + wave) * 16;
  int lr = lane & 15, hi = lane >> 4;

  v8f c0 = vzero8(), c1 = vzero8(), c2 = vzero8(), c3 = vzero8();

#if HAVE_TDM
  extern __shared__ char smem[];
  unsigned ldsbase = (unsigned)(size_t)smem;          // LDS aperture: addr[31:0] = LDS offset
  const char* xblk = (const char*)x + (size_t)blockIdx.x * 64 * NIN * 4;
  const char* wblk = (const char*)WbIn;

  if (tid < 32) {                                     // TDM ignores EXEC; one DMA per block
    tdm_load_2d(ldsbase + 0,     xblk, 2u, 32u, 64u, (unsigned long long)NIN);
    tdm_load_2d(ldsbase + B_OFF, wblk, 3u, 512u, 1u, 512ull);
    __builtin_amdgcn_s_wait_tensorcnt(0);
  }
  __syncthreads();

  for (int kk = 0; kk < 96; kk++) {
    int cb = kk & 1;
    if (kk + 1 < 96 && tid < 32) {
      int nb = (kk + 1) & 1;
      tdm_load_2d(ldsbase + nb * A_BYTES, xblk + (size_t)(kk + 1) * 128,
                  2u, 32u, 64u, (unsigned long long)NIN);
      tdm_load_2d(ldsbase + B_OFF + nb * B_BYTES, wblk + (size_t)(kk + 1) * 4096,
                  3u, 512u, 1u, 512ull);
    }
    const float* ar = (const float*)(smem + cb * A_BYTES) + (wave * 16 + lr) * 32 + (hi ? 8 : 0);
    float4 f0 = *(const float4*)(ar);
    float4 f1 = *(const float4*)(ar + 4);
    float4 f2 = *(const float4*)(ar + 16);
    float4 f3 = *(const float4*)(ar + 20);
    v16h a;
    a[0]  = (h16)f0.x; a[1]  = (h16)f0.y; a[2]  = (h16)f0.z; a[3]  = (h16)f0.w;
    a[4]  = (h16)f1.x; a[5]  = (h16)f1.y; a[6]  = (h16)f1.z; a[7]  = (h16)f1.w;
    a[8]  = (h16)f2.x; a[9]  = (h16)f2.y; a[10] = (h16)f2.z; a[11] = (h16)f2.w;
    a[12] = (h16)f3.x; a[13] = (h16)f3.y; a[14] = (h16)f3.z; a[15] = (h16)f3.w;

    const h16* bp = (const h16*)(smem + B_OFF + cb * B_BYTES) + lane * 16;
    c0 = wmma16(a, load_frag(bp,        bp + 8),        c0);
    c1 = wmma16(a, load_frag(bp + 512,  bp + 512 + 8),  c1);
    c2 = wmma16(a, load_frag(bp + 1024, bp + 1024 + 8), c2);
    c3 = wmma16(a, load_frag(bp + 1536, bp + 1536 + 8), c3);

    if (kk + 1 < 96) {
      if (tid < 32) __builtin_amdgcn_s_wait_tensorcnt(0);
      __syncthreads();
    }
  }
#else
  const float* xr = x + (size_t)(rows + lr) * NIN + (hi ? 8 : 0);
  for (int kk = 0; kk < 96; kk++) {
    const float* px = xr + kk * 32;
    __builtin_prefetch(px + 64, 0, 3);
    float4 f0 = *(const float4*)(px);
    float4 f1 = *(const float4*)(px + 4);
    float4 f2 = *(const float4*)(px + 16);
    float4 f3 = *(const float4*)(px + 20);
    v16h a;
    a[0]  = (h16)f0.x; a[1]  = (h16)f0.y; a[2]  = (h16)f0.z; a[3]  = (h16)f0.w;
    a[4]  = (h16)f1.x; a[5]  = (h16)f1.y; a[6]  = (h16)f1.z; a[7]  = (h16)f1.w;
    a[8]  = (h16)f2.x; a[9]  = (h16)f2.y; a[10] = (h16)f2.z; a[11] = (h16)f2.w;
    a[12] = (h16)f3.x; a[13] = (h16)f3.y; a[14] = (h16)f3.z; a[15] = (h16)f3.w;
    const h16* bp = WbIn + (size_t)kk * 2048 + lane * 16;
    c0 = wmma16(a, load_frag(bp,        bp + 8),        c0);
    c1 = wmma16(a, load_frag(bp + 512,  bp + 512 + 8),  c1);
    c2 = wmma16(a, load_frag(bp + 1024, bp + 1024 + 8), c2);
    c3 = wmma16(a, load_frag(bp + 1536, bp + 1536 + 8), c3);
  }
#endif

  float b0 = biasIn[lr], b1 = biasIn[16 + lr], b2 = biasIn[32 + lr], b3 = biasIn[48 + lr];
#pragma unroll
  for (int i = 0; i < 8; i++) {
    int row = rows + i + hi * 8;
    float* o = actsIn + (size_t)row * 64;
    o[lr]      = fmaxf(c0[i] + b0, 0.f);
    o[16 + lr] = fmaxf(c1[i] + b1, 0.f);
    o[32 + lr] = fmaxf(c2[i] + b2, 0.f);
    o[48 + lr] = fmaxf(c3[i] + b3, 0.f);
  }
}

// ---------------- per-bank: gates + edge WMMAs + relu + partial BN stats ----------------
// grid 64 blocks x 128 rows, block 256 threads (8 waves)

__global__ void k_bank(int t, int nE,
                       int s0, int s1, int s2, int s3,
                       int e0, int e1, int e2, int e3,
                       const h16* __restrict__ actsH, const float* __restrict__ actsIn,
                       const h16* __restrict__ WdF, const float* __restrict__ Wg,
                       const float* __restrict__ bd,
                       float* __restrict__ hbuf, float* psumB, float* psumsqB,
                       float* __restrict__ tgate) {
  __shared__ float gl[4][128];
  __shared__ float hs[128][32];
  int srcs[4] = {s0, s1, s2, s3};
  int eids[4] = {e0, e1, e2, e3};
  int tid = threadIdx.x, blk = blockIdx.x;
  int rbase = blk * 128;

  // gates: gate[e][row] = clip(acts[s][row,:] . Wg[e], 0, 1)
  for (int idx = tid; idx < nE * 128; idx += 256) {
    int ei = idx >> 7, r = idx & 127;
    const h16* ap = actsH + ((size_t)srcs[ei] * NB + rbase + r) * DD;
    const float* wg = Wg + eids[ei] * DD;
    float acc = 0.f;
    for (int k = 0; k < DD; k++) acc += (float)ap[k] * wg[k];
    gl[ei][r] = fminf(fmaxf(acc, 0.f), 1.f);
  }
  __syncthreads();
  if (nE > 0 && tid < 128) {
    float s = 0.f;
    for (int ei = 0; ei < nE; ei++) s += gl[ei][tid];
    tgate[rbase + tid] += s;       // stream-ordered across bank kernels
  }

  int lane = tid & 31, wave = tid >> 5, lr = lane & 15, hi = lane >> 4;
  int rb = rbase + wave * 16;

  v8f a0, a1;
  if (t < 2) {
#pragma unroll
    for (int i = 0; i < 8; i++) {
      int row = rb + i + hi * 8;
      a0[i] = actsIn[(size_t)row * 64 + t * 32 + lr];
      a1[i] = actsIn[(size_t)row * 64 + t * 32 + 16 + lr];
    }
  } else { a0 = vzero8(); a1 = vzero8(); }

  int rowA = rb + lr;
  for (int ei = 0; ei < nE; ei++) {
    int s = srcs[ei], e = eids[ei];
    const h16* ap = actsH + ((size_t)s * NB + rowA) * DD + (hi ? 8 : 0);
    v16h af = load_frag(ap, ap + 16);
    const h16* bp = WdF + e * 1024 + lane * 16;
    v16h bf0 = load_frag(bp,       bp + 8);
    v16h bf1 = load_frag(bp + 512, bp + 512 + 8);
    v8f d0 = wmma16(af, bf0, vzero8());
    v8f d1 = wmma16(af, bf1, vzero8());
    float bd0 = bd[e * DD + lr], bd1 = bd[e * DD + 16 + lr];
#pragma unroll
    for (int i = 0; i < 8; i++) {
      float g = gl[ei][wave * 16 + i + hi * 8];
      a0[i] += g * (d0[i] + bd0);
      a1[i] += g * (d1[i] + bd1);
    }
  }
#pragma unroll
  for (int i = 0; i < 8; i++) {
    int rl = wave * 16 + i + hi * 8;
    hs[rl][lr]      = fmaxf(a0[i], 0.f);
    hs[rl][16 + lr] = fmaxf(a1[i], 0.f);
  }
  __syncthreads();
  if (tid < 32) {
    float s = 0.f, s2 = 0.f;
    for (int r = 0; r < 128; r++) { float v = hs[r][tid]; s += v; s2 += v * v; }
    psumB[blk * 32 + tid] = s;
    psumsqB[blk * 32 + tid] = s2;
  }
  for (int idx = tid; idx < 128 * 32; idx += 256)
    hbuf[(size_t)rbase * 32 + idx] = hs[idx >> 5][idx & 31];
}

// finalize BN (affine=False) and emit f16 acts[t]
__global__ void k_norm(int t, const float* __restrict__ hbuf,
                       const float* psumB, const float* psumsqB, h16* __restrict__ actsH) {
  __shared__ float sm[32], sr[32];
  int tid = threadIdx.x;
  if (tid < 32) {
    float s = 0.f, s2 = 0.f;
    for (int b = 0; b < 64; b++) { s += psumB[b * 32 + tid]; s2 += psumsqB[b * 32 + tid]; }
    float m = s / (float)NB;
    float v = s2 / (float)NB - m * m;
    sm[tid] = m;
    sr[tid] = rsqrtf(v + BN_EPS);
  }
  __syncthreads();
  for (int idx = tid; idx < 128 * 32; idx += 256) {
    size_t g = (size_t)blockIdx.x * 4096 + idx;
    int c = idx & 31;
    actsH[(size_t)t * NB * DD + g] = (h16)((hbuf[g] - sm[c]) * sr[c]);
  }
}

// ---------------- output: acts14 @ Wout0 + acts15 @ Wout1 ----------------
// 512 M-tiles x 63 N-tiles = 32256 tiles, 8 waves/block -> 4032 blocks exactly

__global__ void k_outgemm(const h16* __restrict__ actsH, const h16* __restrict__ WoutF,
                          float* __restrict__ out) {
  int tid = threadIdx.x;
  int lane = tid & 31, wave = tid >> 5;
  int wg = blockIdx.x * 8 + wave;
  int mt = wg / NTOUT, nt = wg % NTOUT;
  int lr = lane & 15, hi = lane >> 4;
  int rowA = mt * 16 + lr;

  const h16* a14p = actsH + ((size_t)14 * NB + rowA) * DD + (hi ? 8 : 0);
  const h16* a15p = actsH + ((size_t)15 * NB + rowA) * DD + (hi ? 8 : 0);
  v16h a14 = load_frag(a14p, a14p + 16);
  v16h a15 = load_frag(a15p, a15p + 16);
  const h16* b14p = WoutF + nt * 512 + lane * 16;
  const h16* b15p = b14p + NTOUT * 512;
  v16h b14 = load_frag(b14p, b14p + 8);
  v16h b15 = load_frag(b15p, b15p + 8);

  v8f c = wmma16(a14, b14, vzero8());
  c = wmma16(a15, b15, c);

  int col = nt * 16 + lr;
  if (col < NOUT) {
#pragma unroll
    for (int i = 0; i < 8; i++)
      out[(size_t)(mt * 16 + i + hi * 8) * NOUT + col] = c[i];
  }
}

// ---------------- host ----------------

extern "C" void kernel_launch(void* const* d_in, const int* in_sizes, int n_in,
                              void* d_out, int out_size, void* d_ws, size_t ws_size,
                              hipStream_t stream) {
  (void)in_sizes; (void)n_in; (void)out_size; (void)ws_size;
  const float* x     = (const float*)d_in[0];
  const float* gamma = (const float*)d_in[1];
  const float* beta  = (const float*)d_in[2];
  const float* Win   = (const float*)d_in[3];
  const float* b_in  = (const float*)d_in[4];
  const float* Wg    = (const float*)d_in[5];
  const float* Wd    = (const float*)d_in[6];
  const float* bd    = (const float*)d_in[7];
  const float* Wout  = (const float*)d_in[8];

  float* out   = (float*)d_out;            // [8192,1000]
  float* tgate = out + (size_t)NB * NOUT;  // [8192]

  // workspace carve-up (all float chunks are multiples of 4 elems -> 16B aligned)
  float* psumIn   = (float*)d_ws;                  // 64*3072
  float* psumsqIn = psumIn + 64 * NIN;             // 64*3072
  float* meanIn   = psumsqIn + 64 * NIN;           // 3072
  float* rstdIn   = meanIn + NIN;                  // 3072
  float* biasIn   = rstdIn + NIN;                  // 64
  float* actsIn   = biasIn + 64;                   // 8192*64
  float* hbuf     = actsIn + (size_t)NB * 64;      // 8192*32
  float* psumB    = hbuf + (size_t)NB * DD;        // 64*32
  float* psumsqB  = psumB + 64 * 32;               // 64*32
  h16*   WbIn     = (h16*)(psumsqB + 64 * 32);     // 3072*64
  h16*   WdF      = WbIn + NIN * 64;               // 54*1024
  h16*   WoutF    = WdF + 54 * 1024;               // 2*63*512
  h16*   actsH    = WoutF + 2 * NTOUT * 512;       // 16*8192*32

  k_zero_tgate<<<32, 256, 0, stream>>>(tgate);
  k_colstats<<<64, 256, 0, stream>>>(x, psumIn, psumsqIn);
  k_colfinal<<<12, 256, 0, stream>>>(psumIn, psumsqIn, meanIn, rstdIn);
  k_fold_win<<<768, 256, 0, stream>>>(Win, gamma, meanIn, rstdIn, WbIn);
  k_fold_bias<<<1, 64, 0, stream>>>(Win, b_in, gamma, beta, meanIn, rstdIn, biasIn);
  k_swz_wd<<<216, 256, 0, stream>>>(Wd, WdF);
  k_swz_wout<<<252, 256, 0, stream>>>(Wout, WoutF);
#if HAVE_TDM
  k_ingemm<<<128, 128, 2 * A_BYTES + 2 * B_BYTES, stream>>>(x, WbIn, biasIn, actsIn);
#else
  k_ingemm<<<128, 128, 0, stream>>>(x, WbIn, biasIn, actsIn);
#endif

  int ebase = 0;
  for (int t = 0; t < 16; t++) {
    int nE = (t < 4) ? t : 4;
    int smin = t - nE;
    int s[4] = {0, 0, 0, 0}, e[4] = {0, 0, 0, 0};
    for (int i = 0; i < nE; i++) { s[i] = smin + i; e[i] = ebase + i; }
    k_bank<<<64, 256, 0, stream>>>(t, nE, s[0], s[1], s[2], s[3],
                                   e[0], e[1], e[2], e[3],
                                   actsH, actsIn, WdF, Wg, bd,
                                   hbuf, psumB, psumsqB, tgate);
    k_norm<<<64, 256, 0, stream>>>(t, hbuf, psumB, psumsqB, actsH);
    ebase += nE;
  }

  k_outgemm<<<4032, 256, 0, stream>>>(actsH, WoutF, out);
}
